// LSTM_OneLayer_78769700209278
// MI455X (gfx1250) — matrix operand order repaired and checked by reading, not compiled, not run on hardware
//
#include <hip/hip_runtime.h>
#include <hip/hip_bf16.h>
#include <stdint.h>

#define EMBED  1024
#define HIDDEN 2048
#define GATES  (4 * HIDDEN)   // 8192
#define LENGTH 4096

// ---- recurrence kernel geometry ----
#define NB     128                  // persistent blocks (grid-wide barrier)
#define TPB    256
#define UPB    (HIDDEN / NB)        // 16 hidden units per block
#define ROWS   (4 * UPB)            // 64 gate rows per block
#define KSPLIT (TPB / ROWS)         // 4 threads per gate row
#define KCHUNK (HIDDEN / KSPLIT)    // 512 K per thread

// ---- workspace layout (all 256-aligned by construction) ----
static constexpr size_t XB_OFF    = 0;                                       // x bf16: 8 MB
static constexpr size_t WIHB_OFF  = XB_OFF    + (size_t)LENGTH * EMBED  * 2; // W_ih bf16: 16 MB
static constexpr size_t WHHB_OFF  = WIHB_OFF  + (size_t)GATES  * EMBED  * 2; // W_hh bf16: 32 MB
static constexpr size_t XPROJ_OFF = WHHB_OFF  + (size_t)GATES  * HIDDEN * 2; // x_proj f32: 128 MB
static constexpr size_t HBUF_OFF  = XPROJ_OFF + (size_t)LENGTH * GATES  * 4; // h: 8 KB
static constexpr size_t SYNC_OFF  = HBUF_OFF  + (size_t)HIDDEN * 4;          // barrier: 2 u32

typedef __attribute__((ext_vector_type(16))) __bf16 v16bf;
typedef __attribute__((ext_vector_type(8)))  __bf16 v8bf;
typedef __attribute__((ext_vector_type(8)))  float  v8f;

union AFrag { v16bf v; v8bf h[2]; };

__device__ __forceinline__ uint16_t f2bf(float f) {
  uint32_t u = __float_as_uint(f);
  return (uint16_t)((u + 0x7FFFu + ((u >> 16) & 1u)) >> 16); // RNE
}
__device__ __forceinline__ float bf_lo(uint32_t u) { return __uint_as_float(u << 16); }
__device__ __forceinline__ float bf_hi(uint32_t u) { return __uint_as_float(u & 0xFFFF0000u); }
__device__ __forceinline__ float sigf(float x) { return 1.0f / (1.0f + __expf(-x)); }

// ---------------------------------------------------------------- convert
__global__ void __launch_bounds__(256) cvt_bf16_kernel(const float* __restrict__ s,
                                                       uint16_t* __restrict__ d, int n4) {
  int i = blockIdx.x * blockDim.x + threadIdx.x;
  if (i < n4) {
    float4 f = ((const float4*)s)[i];
    ushort4 o;
    o.x = f2bf(f.x); o.y = f2bf(f.y); o.z = f2bf(f.z); o.w = f2bf(f.w);
    ((ushort4*)d)[i] = o;
  }
}

// ---------------------------------------------------------------- init state
__global__ void __launch_bounds__(256) init_kernel(float* __restrict__ hbuf,
                                                   unsigned* __restrict__ sync) {
  int i = blockIdx.x * blockDim.x + threadIdx.x;
  if (i < HIDDEN) hbuf[i] = 0.0f;
  if (i < 2) sync[i] = 0u;
}

// ---------------------------------------------------------------- x_proj GEMM (WMMA bf16)
// out[t][j] = sum_k x[t][k]*W_ih[j][k] + b_ih[j] + b_hh[j]
// block = 8 waves -> 32x256 tile; each wave: one 16-row strip x four 16-col tiles.
__global__ void __launch_bounds__(256) xproj_gemm_kernel(
    const uint16_t* __restrict__ Xb,   // [LENGTH][EMBED] bf16
    const uint16_t* __restrict__ Wb,   // [GATES][EMBED] bf16
    const float* __restrict__ b_ih, const float* __restrict__ b_hh,
    float* __restrict__ out)           // [LENGTH][GATES] f32
{
  const int lane = threadIdx.x & 31;
  const int wave = threadIdx.x >> 5;
  const int wm = wave >> 2, wn = wave & 3;
  const int mBase = blockIdx.y * 32 + wm * 16;
  const int nBase = blockIdx.x * 256 + wn * 64;
  const int sel = lane >> 4;   // 0/1: K-half select (contiguous k = 16*sel + s, same for A and B)
  const int l15 = lane & 15;

  const uint16_t* arow = Xb + (size_t)(mBase + l15) * EMBED;

  v8f acc[4] = {v8f{}, v8f{}, v8f{}, v8f{}};

  for (int k = 0; k < EMBED; k += 32) {
    AFrag A;
    A.h[0] = *(const v8bf*)(const void*)(arow + k + sel * 16);       // K = base + sel*16 .. +7
    A.h[1] = *(const v8bf*)(const void*)(arow + k + sel * 16 + 8);   // K = base + sel*16+8 ..
#pragma unroll
    for (int tI = 0; tI < 4; ++tI) {
      const v16bf B = *(const v16bf*)(const void*)
          (Wb + (size_t)(nBase + tI * 16 + l15) * EMBED + k + sel * 16);
      acc[tI] = __builtin_amdgcn_wmma_f32_16x16x32_bf16(
          /*neg_a=*/false, A.v, /*neg_b=*/false, B,
          /*c_mod=*/(short)0, acc[tI], /*reuse_a=*/false, /*reuse_b=*/false);
    }
  }

#pragma unroll
  for (int tI = 0; tI < 4; ++tI) {
    const int n = nBase + tI * 16 + l15;
    const float bias = b_ih[n] + b_hh[n];
#pragma unroll
    for (int r = 0; r < 8; ++r) {
      const int m = mBase + sel * 8 + r;     // C/D layout: vgpr r, lane-half sel
      out[(size_t)m * GATES + n] = acc[tI][r] + bias;
    }
  }
}

// ---------------------------------------------------------------- grid barrier
__device__ __forceinline__ void grid_barrier(unsigned* cnt, unsigned* gen, unsigned nblocks) {
  __syncthreads();
  if (threadIdx.x == 0) {
    __threadfence();  // release this block's h writes to device scope
    unsigned g = __hip_atomic_load(gen, __ATOMIC_RELAXED, __HIP_MEMORY_SCOPE_AGENT);
    unsigned a = __hip_atomic_fetch_add(cnt, 1u, __ATOMIC_ACQ_REL, __HIP_MEMORY_SCOPE_AGENT);
    if (a == nblocks - 1u) {
      __hip_atomic_store(cnt, 0u, __ATOMIC_RELAXED, __HIP_MEMORY_SCOPE_AGENT);
      __hip_atomic_fetch_add(gen, 1u, __ATOMIC_RELEASE, __HIP_MEMORY_SCOPE_AGENT);
    } else {
      while (__hip_atomic_load(gen, __ATOMIC_ACQUIRE, __HIP_MEMORY_SCOPE_AGENT) == g)
        __builtin_amdgcn_s_sleep(1);
    }
    __threadfence();
  }
  __syncthreads();
}

// ---------------------------------------------------------------- persistent LSTM scan
__global__ void __launch_bounds__(TPB) lstm_scan_kernel(
    const float* __restrict__ xproj,     // [LENGTH][GATES]
    const uint16_t* __restrict__ whh,    // [GATES][HIDDEN] bf16 (L2-resident, 32 MB)
    float* __restrict__ hbuf,            // [HIDDEN]
    unsigned* __restrict__ sync)         // [2]: count, generation
{
  __shared__ float sh[HIDDEN];           // staged h (8 KB)
  __shared__ float sg[ROWS];             // per-block gate values

  const int tid = threadIdx.x;
  const int row = tid >> 2;              // 0..63 gate row within block
  const int ts  = tid & 3;               // K-split lane
  const int q   = row / UPB;             // gate: 0=i 1=f 2=g 3=o
  const int u   = row % UPB;             // hidden unit within block
  const int gi  = q * HIDDEN + (int)blockIdx.x * UPB + u;   // global gate row
  const uint16_t* wrow = whh + (size_t)gi * HIDDEN + ts * KCHUNK;
  const int kb = ts * KCHUNK;

  float c = 0.0f;                        // cell state lives in registers (tid < UPB)

  for (int t = 0; t < LENGTH; ++t) {
    // stage h into LDS (coalesced)
#pragma unroll
    for (int j = 0; j < HIDDEN / TPB; ++j)
      sh[tid + j * TPB] = hbuf[tid + j * TPB];
    __syncthreads();

    // GEMV partial: 512 MACs per thread, bf16 weights unpacked to f32
    float p = 0.0f;
#pragma unroll 4
    for (int k = 0; k < KCHUNK; k += 8) {
      uint4 w = *(const uint4*)(const void*)(wrow + k);
      const float* hv = &sh[kb + k];
      p = fmaf(bf_lo(w.x), hv[0], p); p = fmaf(bf_hi(w.x), hv[1], p);
      p = fmaf(bf_lo(w.y), hv[2], p); p = fmaf(bf_hi(w.y), hv[3], p);
      p = fmaf(bf_lo(w.z), hv[4], p); p = fmaf(bf_hi(w.z), hv[5], p);
      p = fmaf(bf_lo(w.w), hv[6], p); p = fmaf(bf_hi(w.w), hv[7], p);
    }
    // reduce the 4 K-split partials (lanes are consecutive within the wave)
    p += __shfl_xor(p, 1);
    p += __shfl_xor(p, 2);
    if (ts == 0) sg[row] = p + xproj[(size_t)t * GATES + gi];
    if (t + 1 < LENGTH)
      __builtin_prefetch(&xproj[(size_t)(t + 1) * GATES + gi], 0, 1); // global_prefetch_b8
    __syncthreads();

    if (tid < UPB) {
      float iv = sigf (sg[tid]);
      float fv = sigf (sg[UPB + tid]);
      float gv = tanhf(sg[2 * UPB + tid]);
      float ov = sigf (sg[3 * UPB + tid]);
      c = fv * c + iv * gv;
      hbuf[(int)blockIdx.x * UPB + tid] = ov * tanhf(c);
    }
    grid_barrier(sync, sync + 1, gridDim.x);
  }
}

// ---------------------------------------------------------------- FC head
__global__ void __launch_bounds__(256) fc_kernel(const float* __restrict__ h,
                                                 const float* __restrict__ wfc,
                                                 const float* __restrict__ bfc,
                                                 float* __restrict__ out) {
  __shared__ float red[256];
  float s = 0.0f;
  for (int k = threadIdx.x; k < HIDDEN; k += 256) s += h[k] * wfc[k];
  red[threadIdx.x] = s;
  __syncthreads();
  for (int off = 128; off > 0; off >>= 1) {
    if (threadIdx.x < off) red[threadIdx.x] += red[threadIdx.x + off];
    __syncthreads();
  }
  if (threadIdx.x == 0) out[0] = sigf(red[0] + bfc[0]);
}

// ---------------------------------------------------------------- launch
extern "C" void kernel_launch(void* const* d_in, const int* in_sizes, int n_in,
                              void* d_out, int out_size, void* d_ws, size_t ws_size,
                              hipStream_t stream) {
  const float* x   = (const float*)d_in[0];
  const float* Wih = (const float*)d_in[1];
  const float* Whh = (const float*)d_in[2];
  const float* bih = (const float*)d_in[3];
  const float* bhh = (const float*)d_in[4];
  const float* Wfc = (const float*)d_in[5];
  const float* bfc = (const float*)d_in[6];
  float* out = (float*)d_out;

  char* ws = (char*)d_ws;
  uint16_t* xb   = (uint16_t*)(ws + XB_OFF);
  uint16_t* wihb = (uint16_t*)(ws + WIHB_OFF);
  uint16_t* whhb = (uint16_t*)(ws + WHHB_OFF);
  float*    xp   = (float*)   (ws + XPROJ_OFF);
  float*    hbuf = (float*)   (ws + HBUF_OFF);
  unsigned* sync = (unsigned*)(ws + SYNC_OFF);

  // fp32 -> bf16 copies (4 elems/thread)
  {
    int n4 = (LENGTH * EMBED) / 4;
    cvt_bf16_kernel<<<(n4 + 255) / 256, 256, 0, stream>>>(x, xb, n4);
    n4 = (GATES * EMBED) / 4;
    cvt_bf16_kernel<<<(n4 + 255) / 256, 256, 0, stream>>>(Wih, wihb, n4);
    n4 = (GATES * HIDDEN) / 4;
    cvt_bf16_kernel<<<(n4 + 255) / 256, 256, 0, stream>>>(Whh, whhb, n4);
  }

  // zero h0 and barrier state (ws is poisoned, must init every call)
  init_kernel<<<(HIDDEN + 255) / 256, 256, 0, stream>>>(hbuf, sync);

  // phase 1: x_proj = x @ W_ih.T + b_ih + b_hh   (WMMA bf16 -> f32)
  dim3 ggrid(GATES / 256, LENGTH / 32);
  xproj_gemm_kernel<<<ggrid, 256, 0, stream>>>(xb, wihb, bih, bhh, xp);

  // phase 2: persistent sequential scan with grid-wide barrier per step
  lstm_scan_kernel<<<NB, TPB, 0, stream>>>(xp, whhb, hbuf, sync);

  // phase 3: sigmoid(h @ W_fc.T + b_fc)
  fc_kernel<<<1, 256, 0, stream>>>(hbuf, Wfc, bfc, out);
}
